// TopKBase_26955214750255
// MI455X (gfx1250) — matrix operand order repaired and checked by reading, not compiled, not run on hardware
//
#include <hip/hip_runtime.h>

typedef __attribute__((ext_vector_type(16))) __bf16 v16bf;
typedef __attribute__((ext_vector_type(8)))  float  v8f;
typedef __attribute__((ext_vector_type(2)))  __bf16 v2bf;

#define ROWS     4096     // B*T
#define EDIM     768
#define FDIM     16384
#define KTOP     32
#define TILE_M   128
#define TILE_N   128
#define TILE_K   32
#define LDSS     40       // ushorts per LDS row (32 + 8 pad) -> 80B rows, 16B-aligned, 20-dword bank stride

struct alignas(16) U4 { unsigned int x, y, z, w; };
struct alignas(16) F4 { float x, y, z, w; };
struct alignas(8)  U2 { unsigned int x, y; };

__device__ __forceinline__ unsigned short f32_bf16(float f) {
    unsigned int u = __builtin_bit_cast(unsigned int, f);
    u += 0x7fffu + ((u >> 16) & 1u);           // round-to-nearest-even
    return (unsigned short)(u >> 16);
}

__device__ __forceinline__ unsigned int pack2(float a, float b) {
#if __has_builtin(__builtin_amdgcn_cvt_pk_bf16_f32)
    v2bf p = __builtin_amdgcn_cvt_pk_bf16_f32(a, b);
    return __builtin_bit_cast(unsigned int, p);
#else
    return (unsigned int)f32_bf16(a) | ((unsigned int)f32_bf16(b) << 16);
#endif
}

// ---------------------------------------------------------------------------
// Kernel 1: latent = relu((x - b_dec) @ W_enc + b_enc), bf16 WMMA, f32 accum.
// Writes latent into the feature_magnitudes region of d_out.
// ---------------------------------------------------------------------------
__global__ __launch_bounds__(256)
void sae_encoder_wmma(const float* __restrict__ x,
                      const float* __restrict__ W_enc,
                      const float* __restrict__ b_enc,
                      const float* __restrict__ b_dec,
                      float* __restrict__ latent) {
    __shared__ unsigned short As[TILE_M * LDSS];   // [m][k] bf16
    __shared__ unsigned short Bs[TILE_N * LDSS];   // [n][k] bf16 (transposed W_enc tile)

    const int t      = threadIdx.x;
    const int lane   = t & 31;
    const int wave   = t >> 5;                 // 0..7
    const int nBlock = blockIdx.x * TILE_N;    // feature offset
    const int mBlock = blockIdx.y * TILE_M;    // token-row offset

    const int wm = (wave & 3) * 32;            // wave's 32-row sub-tile (2 x 16)
    const int wn = (wave >> 2) * 64;           // wave's 64-col sub-tile (4 x 16)

    v8f acc[2][4] = {};

    const int mRow   = lane & 15;
    const int kGroup = lane >> 4;              // 0 or 1

    // staging coordinates (loop-invariant)
    const int ac4 = (t & 7) * 4;               // A: k quad
    const int ar0 = t >> 3;                    // A: row 0..31
    const int bn  = t & 127;                   // B: feature col
    const int bkq = (t >> 7) * 4;              // B: k quad base (0 or 4)

    for (int kT = 0; kT < EDIM; kT += TILE_K) {
        // ---- stage A tile: (x - b_dec)[mBlock..+128][kT..+32] as bf16 ----
        {
            const F4 bd4 = *reinterpret_cast<const F4*>(b_dec + kT + ac4);
            #pragma unroll
            for (int i = 0; i < 4; i++) {
                const int r = ar0 + i * 32;
                const F4 a = *reinterpret_cast<const F4*>(
                    x + (size_t)(mBlock + r) * EDIM + kT + ac4);
                U2 p;
                p.x = pack2(a.x - bd4.x, a.y - bd4.y);
                p.y = pack2(a.z - bd4.z, a.w - bd4.w);
                *reinterpret_cast<U2*>(&As[r * LDSS + ac4]) = p;
            }
        }
        // ---- stage B tile: W_enc[kT..+32][nBlock..+128], stored transposed ----
        {
            #pragma unroll
            for (int i = 0; i < 4; i++) {
                const int kk = bkq + i * 8;    // 4 consecutive k per store
                const size_t base = (size_t)(kT + kk) * FDIM + nBlock + bn;
                const float w0 = W_enc[base];
                const float w1 = W_enc[base + FDIM];
                const float w2 = W_enc[base + 2 * (size_t)FDIM];
                const float w3 = W_enc[base + 3 * (size_t)FDIM];
                U2 p;
                p.x = pack2(w0, w1);
                p.y = pack2(w2, w3);
                *reinterpret_cast<U2*>(&Bs[bn * LDSS + kk]) = p;
            }
        }
        __syncthreads();

        // ---- assemble fragments per ISA 16-bit A(16x32) / B(32x16) layouts ----
        // A: lanes 0-15 (g=0): V0-3 = K0..7, V4-7 = K16..23; lanes 16-31 (g=1): +8
        //    -> two contiguous 16B chunks: rowp[g] and rowp[2+g]
        // B: same K order as A: lanes 0-15: K0..7, K16..23; lanes 16-31: +8
        //    -> two contiguous 16B chunks: colp[g] and colp[2+g]
        v16bf aF[2], bF[4];
        #pragma unroll
        for (int i = 0; i < 2; i++) {
            const U4* rowp = reinterpret_cast<const U4*>(
                &As[(wm + i * 16 + mRow) * LDSS]);
            union { v16bf v; U4 q[2]; } fa;
            fa.q[0] = rowp[kGroup];
            fa.q[1] = rowp[2 + kGroup];
            aF[i] = fa.v;
        }
        #pragma unroll
        for (int j = 0; j < 4; j++) {
            const U4* colp = reinterpret_cast<const U4*>(
                &Bs[(wn + j * 16 + mRow) * LDSS]);
            union { v16bf v; U4 q[2]; } fb;
            fb.q[0] = colp[kGroup];
            fb.q[1] = colp[2 + kGroup];
            bF[j] = fb.v;
        }

        #pragma unroll
        for (int i = 0; i < 2; i++)
            #pragma unroll
            for (int j = 0; j < 4; j++)
                acc[i][j] = __builtin_amdgcn_wmma_f32_16x16x32_bf16(
                    false, aF[i], false, bF[j], (short)0, acc[i][j], false, false);

        __syncthreads();
    }

    // ---- epilogue: + b_enc, ReLU, store f32 ----
    // C/D layout: vgpr r -> row (r + 8*(lane>>4)), col (lane & 15)
    const int cRow = 8 * kGroup;
    const int cCol = mRow;
    #pragma unroll
    for (int j = 0; j < 4; j++) {
        const int n = nBlock + wn + j * 16 + cCol;
        const float be = b_enc[n];
        #pragma unroll
        for (int i = 0; i < 2; i++) {
            const int mBase = mBlock + wm + i * 16 + cRow;
            #pragma unroll
            for (int r = 0; r < 8; r++) {
                float val = acc[i][j][r] + be;
                val = val > 0.0f ? val : 0.0f;
                latent[(size_t)(mBase + r) * FDIM + n] = val;
            }
        }
    }
}

// ---------------------------------------------------------------------------
// Kernel 2: per-row exact top-32 (ties -> lower index, like jax.lax.top_k),
// threshold-mask in place, emit indices, sparse decode via W_dec gather.
// ---------------------------------------------------------------------------
__global__ __launch_bounds__(256)
void sae_topk_decode(float* __restrict__ feat,       // latent in, masked out (in place)
                     float* __restrict__ xrec,
                     int*   __restrict__ idx_out,
                     const float* __restrict__ W_dec,
                     const float* __restrict__ b_dec) {
    const int row = blockIdx.x;
    const int t   = threadIdx.x;
    float* rp = feat + (size_t)row * FDIM;

    // element e owned by thread e%256, slot e/256 (coalesced loads)
    float v[64];
    #pragma unroll
    for (int s = 0; s < 64; s++) v[s] = rp[s * 256 + t];

    float lmax = v[0]; int ls = 0;
    #pragma unroll
    for (int s = 1; s < 64; s++)
        if (v[s] > lmax) { lmax = v[s]; ls = s; }   // ascending s => lowest index wins ties

    __shared__ float sv[9];
    __shared__ int   si[9];
    __shared__ float topv[KTOP];
    __shared__ int   topi[KTOP];

    for (int it = 0; it < KTOP; it++) {
        // wave32 argmax via shuffles (no LDS, no barriers)
        float bv = lmax;
        int   bi = ls * 256 + t;
        #pragma unroll
        for (int m = 16; m > 0; m >>= 1) {
            const float ov = __shfl_xor(bv, m, 32);
            const int   oi = __shfl_xor(bi, m, 32);
            if (ov > bv || (ov == bv && oi < bi)) { bv = ov; bi = oi; }
        }
        if ((t & 31) == 0) { sv[t >> 5] = bv; si[t >> 5] = bi; }
        __syncthreads();
        if (t == 0) {
            float w = sv[0]; int wi0 = si[0];
            #pragma unroll
            for (int q = 1; q < 8; q++)
                if (sv[q] > w || (sv[q] == w && si[q] < wi0)) { w = sv[q]; wi0 = si[q]; }
            sv[8] = w; si[8] = wi0;
            topv[it] = w; topi[it] = wi0;
        }
        __syncthreads();
        const int wi = si[8];
        if ((wi & 255) == t) {                       // owner removes winner, rescans
            v[wi >> 8] = -1.0f;
            lmax = v[0]; ls = 0;
            #pragma unroll
            for (int s = 1; s < 64; s++)
                if (v[s] > lmax) { lmax = v[s]; ls = s; }
        }
    }

    const float thr = topv[KTOP - 1];

    // restore winners, then threshold-mask write-back (latent >= thr kept)
    #pragma unroll
    for (int j = 0; j < KTOP; j++) {
        const int wi = topi[j];
        if ((wi & 255) == t) v[wi >> 8] = topv[j];
    }
    #pragma unroll
    for (int s = 0; s < 64; s++) {
        const float val = v[s];
        rp[s * 256 + t] = (val >= thr) ? val : 0.0f;
    }

    if (t < KTOP) idx_out[row * KTOP + t] = topi[t];

    // sparse decode: x_rec[row] = sum_j topv[j] * W_dec[topi[j], :] + b_dec
    #pragma unroll
    for (int e = t; e < EDIM; e += 256) {
        float acc = b_dec[e];
        #pragma unroll 8
        for (int j = 0; j < KTOP; j++)
            acc += topv[j] * W_dec[(size_t)topi[j] * EDIM + e];
        xrec[(size_t)row * EDIM + e] = acc;
    }
}

// ---------------------------------------------------------------------------
extern "C" void kernel_launch(void* const* d_in, const int* in_sizes, int n_in,
                              void* d_out, int out_size, void* d_ws, size_t ws_size,
                              hipStream_t stream) {
    (void)in_sizes; (void)n_in; (void)out_size; (void)d_ws; (void)ws_size;

    const float* x     = (const float*)d_in[0];
    const float* W_enc = (const float*)d_in[1];
    const float* b_enc = (const float*)d_in[2];
    const float* W_dec = (const float*)d_in[3];
    const float* b_dec = (const float*)d_in[4];
    // d_in[5] is k (==32), baked in as KTOP

    float* xrec = (float*)d_out;                                // [4096 x 768]
    float* feat = xrec + (size_t)ROWS * EDIM;                   // [4096 x 16384]
    int*   idxo = (int*)(feat + (size_t)ROWS * FDIM);           // [4096 x 32]

    dim3 g1(FDIM / TILE_N, ROWS / TILE_M);                      // (128, 32)
    sae_encoder_wmma<<<g1, 256, 0, stream>>>(x, W_enc, b_enc, b_dec, feat);
    sae_topk_decode<<<ROWS, 256, 0, stream>>>(feat, xrec, idxo, W_dec, b_dec);
}
